// PromptGeneratorWithPooling_28656021799561
// MI455X (gfx1250) — matrix operand order repaired and checked by reading, not compiled, not run on hardware
//
#include <hip/hip_runtime.h>

#define BB   8
#define SS   4096
#define DD   1024
#define CC   256
#define PP   64
#define WIN  64          // SS / PP
#define ROWP 1032        // padded LDS row length in bf16 elems (2064 B: 16-B bank step)

typedef __attribute__((ext_vector_type(16))) __bf16 v16bf;
typedef __attribute__((ext_vector_type(8)))  float  v8f;

struct Pack32 { uint4 lo, hi; };   // 32 bytes == one v16bf fragment

__device__ __forceinline__ unsigned short f2bf(float f) {
    unsigned u = __float_as_uint(f);
    u += 0x7FFFu + ((u >> 16) & 1u);          // round-to-nearest-even
    return (unsigned short)(u >> 16);
}

__device__ __forceinline__ v16bf frag2(const unsigned short* p0, const unsigned short* p1) {
    Pack32 p;
    p.lo = *(const uint4*)p0;
    p.hi = *(const uint4*)p1;
    return __builtin_bit_cast(v16bf, p);
}

// ---------------------------------------------------------------- kernel 0
// Convert W_down [C,D] and W_up [D,C] to bf16 in workspace (both 256K elems).
__global__ void __launch_bounds__(256)
k_cvt_weights(const float* __restrict__ Wd, const float* __restrict__ Wu,
              unsigned short* __restrict__ wd_bf, unsigned short* __restrict__ wu_bf) {
    int i = blockIdx.x * 256 + threadIdx.x;
    if (i < CC * DD) {
        wd_bf[i] = f2bf(Wd[i]);
        wu_bf[i] = f2bf(Wu[i]);
    }
}

// ---------------------------------------------------------------- kernel 1
// One workgroup per (b,p) window: GEMM 64x1024 * 1024x256 with WMMA bf16,
// fused windowed max-pool + bias + ReLU in the accumulator epilogue.
// All six fragments of a k-step live in distinct registers before any WMMA
// issues, and the k-loop is unrolled x2, so the compiler can pipeline the
// next step's ds/global loads under the current WMMAs instead of inserting
// WAR-hazard v_nops.
__global__ void __launch_bounds__(256)
k_down_pool(const float* __restrict__ X, const unsigned short* __restrict__ wd,
            const float* __restrict__ b_down, float* __restrict__ pooled) {
    extern __shared__ unsigned short sX[];     // [WIN][ROWP] bf16, 132096 B
    const int t    = threadIdx.x;
    const int wave = t >> 5;
    const int lane = t & 31;
    const int l16  = lane & 15;
    const bool hi  = lane >= 16;

    // ---- stage window (64 x 1024 f32) -> LDS bf16; one row per iteration,
    //      256 threads * float4 == exactly one 1024-col row, conflict-free.
    const float4* src = (const float4*)(X + (size_t)blockIdx.x * (WIN * DD));
    for (int r = 0; r < WIN; ++r) {
        float4 f = src[r * (DD / 4) + t];
        uint2 pk;
        pk.x = (unsigned)f2bf(f.x) | ((unsigned)f2bf(f.y) << 16);
        pk.y = (unsigned)f2bf(f.z) | ((unsigned)f2bf(f.w) << 16);
        *(uint2*)&sX[r * ROWP + t * 4] = pk;
    }
    __syncthreads();

    // ---- wave owns channels [wave*32, wave*32+32) for all 4 M-tiles
    v8f acc[4][2] = {};
    const int cBase = wave * 32;
    const int aOff = hi ? 8 : 0;     // A: lanes>=16 hold K 8..15 / 24..31
    const int bOff = hi ? 8 : 0;     // B: lanes>=16 hold K 8..15 / 24..31

    const unsigned short* bcol0 = wd + (size_t)(cBase + l16) * DD + bOff;       // n-tile 0
    const unsigned short* bcol1 = wd + (size_t)(cBase + 16 + l16) * DD + bOff;  // n-tile 1
    const unsigned short* aRow0 = &sX[( 0 + l16) * ROWP] + aOff;
    const unsigned short* aRow1 = &sX[(16 + l16) * ROWP] + aOff;
    const unsigned short* aRow2 = &sX[(32 + l16) * ROWP] + aOff;
    const unsigned short* aRow3 = &sX[(48 + l16) * ROWP] + aOff;

#pragma unroll 2
    for (int k0 = 0; k0 < DD; k0 += 32) {
        // --- all fragments of this k-step in distinct registers
        v16bf B0 = frag2(bcol0 + k0,      bcol0 + k0 + 16);
        v16bf B1 = frag2(bcol1 + k0,      bcol1 + k0 + 16);
        v16bf A0 = frag2(aRow0 + k0,      aRow0 + k0 + 16);
        v16bf A1 = frag2(aRow1 + k0,      aRow1 + k0 + 16);
        v16bf A2 = frag2(aRow2 + k0,      aRow2 + k0 + 16);
        v16bf A3 = frag2(aRow3 + k0,      aRow3 + k0 + 16);
        // --- 8 independent WMMAs (distinct accumulators, no RAW chains)
        acc[0][0] = __builtin_amdgcn_wmma_f32_16x16x32_bf16(
            false, A0, false, B0, (short)0, acc[0][0], false, false);
        acc[1][0] = __builtin_amdgcn_wmma_f32_16x16x32_bf16(
            false, A1, false, B0, (short)0, acc[1][0], false, false);
        acc[2][0] = __builtin_amdgcn_wmma_f32_16x16x32_bf16(
            false, A2, false, B0, (short)0, acc[2][0], false, false);
        acc[3][0] = __builtin_amdgcn_wmma_f32_16x16x32_bf16(
            false, A3, false, B0, (short)0, acc[3][0], false, false);
        acc[0][1] = __builtin_amdgcn_wmma_f32_16x16x32_bf16(
            false, A0, false, B1, (short)0, acc[0][1], false, false);
        acc[1][1] = __builtin_amdgcn_wmma_f32_16x16x32_bf16(
            false, A1, false, B1, (short)0, acc[1][1], false, false);
        acc[2][1] = __builtin_amdgcn_wmma_f32_16x16x32_bf16(
            false, A2, false, B1, (short)0, acc[2][1], false, false);
        acc[3][1] = __builtin_amdgcn_wmma_f32_16x16x32_bf16(
            false, A3, false, B1, (short)0, acc[3][1], false, false);
    }

    // ---- epilogue: max over all 64 rows, + b_down, ReLU, store [512,256]
#pragma unroll
    for (int nt = 0; nt < 2; ++nt) {
        v8f m = acc[0][nt];
#pragma unroll
        for (int mt = 1; mt < 4; ++mt)
#pragma unroll
            for (int r = 0; r < 8; ++r) m[r] = fmaxf(m[r], acc[mt][nt][r]);
        float v = m[0];
#pragma unroll
        for (int r = 1; r < 8; ++r) v = fmaxf(v, m[r]);
        v = fmaxf(v, __shfl_xor(v, 16, 32));           // merge M-halves (same column)
        const int c = cBase + nt * 16 + l16;
        v = fmaxf(v + b_down[c], 0.0f);
        if (!hi) pooled[(size_t)blockIdx.x * CC + c] = v;
    }
}

// ---------------------------------------------------------------- kernel 2
// BatchNorm stats over the 512 pooled rows (one thread per channel), then
// write normalized activations as bf16 for the up-proj GEMM.
__global__ void __launch_bounds__(256)
k_bnstats(const float* __restrict__ pooled, const float* __restrict__ gamma,
          const float* __restrict__ beta, unsigned short* __restrict__ pooledN) {
    const int c = threadIdx.x;
    float s = 0.f, s2 = 0.f;
    for (int r = 0; r < BB * PP; ++r) {
        float v = pooled[r * CC + c];
        s += v; s2 += v * v;
    }
    const float inv  = 1.0f / (float)(BB * PP);
    const float mean = s * inv;
    const float var  = s2 * inv - mean * mean;     // biased, as in BN
    const float scale = gamma[c] * rsqrtf(var + 1e-5f);
    const float shift = beta[c] - mean * scale;
    for (int r = 0; r < BB * PP; ++r)
        pooledN[r * CC + c] = f2bf(pooled[r * CC + c] * scale + shift);
}

// ---------------------------------------------------------------- kernel 3
// Up-proj: [512,256] x [256,1024] WMMA bf16 GEMM + b_up, f32 out.
__global__ void __launch_bounds__(256)
k_up(const unsigned short* __restrict__ pooledN, const unsigned short* __restrict__ wu,
     const float* __restrict__ b_up, float* __restrict__ out) {
    const int t    = threadIdx.x;
    const int wave = t >> 5;
    const int lane = t & 31;
    const int l16  = lane & 15;
    const bool hi  = lane >= 16;
    const int m0 = blockIdx.x * 16;
    const int n0 = blockIdx.y * 128 + wave * 16;

    v8f acc = {};
    const int aOff = hi ? 8 : 0;
    const int bOff = hi ? 8 : 0;
    const unsigned short* arow = pooledN + (size_t)(m0 + l16) * CC + aOff;
    const unsigned short* bcol = wu + (size_t)(n0 + l16) * CC + bOff;  // W_up row n == B col n

#pragma unroll 2
    for (int k0 = 0; k0 < CC; k0 += 32) {
        v16bf A = frag2(arow + k0, arow + k0 + 16);
        v16bf B = frag2(bcol + k0, bcol + k0 + 16);
        acc = __builtin_amdgcn_wmma_f32_16x16x32_bf16(
            false, A, false, B, (short)0, acc, false, false);
    }

    const int n = n0 + l16;
    const float bias = b_up[n];
    const int mb = m0 + (hi ? 8 : 0);
#pragma unroll
    for (int r = 0; r < 8; ++r)
        out[(size_t)(mb + r) * DD + n] = acc[r] + bias;
}

// ---------------------------------------------------------------- launch
extern "C" void kernel_launch(void* const* d_in, const int* in_sizes, int n_in,
                              void* d_out, int out_size, void* d_ws, size_t ws_size,
                              hipStream_t stream) {
    const float* X     = (const float*)d_in[0];
    const float* Wd    = (const float*)d_in[1];
    const float* bdown = (const float*)d_in[2];
    const float* gamma = (const float*)d_in[3];
    const float* beta  = (const float*)d_in[4];
    const float* Wu    = (const float*)d_in[5];
    const float* bup   = (const float*)d_in[6];
    float* out = (float*)d_out;

    char* ws = (char*)d_ws;
    unsigned short* wd_bf   = (unsigned short*)(ws);              // 512 KB
    unsigned short* wu_bf   = (unsigned short*)(ws + 524288);     // 512 KB
    float*          pooled  = (float*)(ws + 1048576);             // 512 KB
    unsigned short* pooledN = (unsigned short*)(ws + 1572864);    // 256 KB

    k_cvt_weights<<<(CC * DD + 255) / 256, 256, 0, stream>>>(Wd, Wu, wd_bf, wu_bf);
    k_down_pool<<<BB * PP, 256, WIN * ROWP * sizeof(unsigned short), stream>>>(
        X, wd_bf, bdown, pooled);
    k_bnstats<<<1, 256, 0, stream>>>(pooled, gamma, beta, pooledN);
    k_up<<<dim3(BB * PP / 16, DD / 128), 256, 0, stream>>>(pooledN, wu_bf, bup, out);
}
